// BipartiteGNN_5454608466090
// MI455X (gfx1250) — matrix operand order repaired and checked by reading, not compiled, not run on hardware
//
#include <hip/hip_runtime.h>

typedef __attribute__((ext_vector_type(16))) __bf16   v16bf;
typedef __attribute__((ext_vector_type(8)))  float    v8f;
typedef __attribute__((ext_vector_type(4)))  unsigned uint4v;

#define DD 64  // hidden dim (fixed by the reference)

// ---------- helpers ----------

__device__ __forceinline__ __bf16 f2bf(float f) {
  unsigned u;
  __builtin_memcpy(&u, &f, 4);
  u += 0x7FFFu + ((u >> 16) & 1u);   // round-to-nearest-even
  unsigned short h = (unsigned short)(u >> 16);
  __bf16 r;
  __builtin_memcpy(&r, &h, 2);
  return r;
}

__device__ __forceinline__ float bf2f(__bf16 b) {
  unsigned short h;
  __builtin_memcpy(&h, &b, 2);
  unsigned u = ((unsigned)h) << 16;
  float f;
  __builtin_memcpy(&f, &u, 4);
  return f;
}

__device__ __forceinline__ float relu(float x) {
  return __builtin_amdgcn_fmed3f(x, 0.0f, 3.402823466e38f);  // v_med3_f32
}

// Dense 16-bit A-matrix 16x32 layout (wave32): lane (0..15) holds row M=lane.
// element i -> K = kb + (i&7) + 16*(i>>3) + 8*half
union AFrag { uint4v u[2]; v16bf v; };

// base is wave-uniform; elem is a per-lane u32 element offset -> SADDR+voffset form.
__device__ __forceinline__ v16bf load_a_bf(const __bf16* __restrict__ base, unsigned elem,
                                           int kb, int half) {
  unsigned o = elem + (unsigned)(kb + 8 * half);
  AFrag f;
  f.u[0] = *(const uint4v*)(base + o);
  f.u[1] = *(const uint4v*)(base + o + 16);
  return f.v;
}

__device__ __forceinline__ v16bf load_a_f32(const float* __restrict__ base, unsigned elem,
                                            int kb, int half) {
  const float* p0 = base + elem + (unsigned)(kb + 8 * half);
  v16bf a;
#pragma unroll
  for (int i = 0; i < 8; ++i) {
    a[i]     = f2bf(p0[i]);
    a[i + 8] = f2bf(p0[i + 16]);
  }
  return a;
}

// Cooperative copy of packed W fragments into LDS; includes the barrier.
__device__ __forceinline__ void stage_w(const __bf16* __restrict__ Wp, __bf16* lds, int nfrags) {
  const int chunks = nfrags * 64;  // 16-byte chunks
  for (int i = threadIdx.x; i < chunks; i += blockDim.x)
    ((uint4v*)lds)[i] = ((const uint4v*)Wp)[i];
  __syncthreads();
}

// Fragment read from LDS. Volatile (so it is re-read per use, never hoisted/spilled)
// AND explicitly address_space(3) so it lowers to ds_load_b128 (InferAddressSpaces
// skips volatile ops, which would otherwise produce flat SCOPE_SYS loads).
typedef __attribute__((address_space(3))) const volatile uint4v* lds_frag_p;
__device__ __forceinline__ v16bf load_b_lds(const __bf16* lds, int f, int lane) {
  lds_frag_p p = (lds_frag_p)(lds + f * 512 + lane * 16);
  AFrag r;
  r.u[0] = p[0];
  r.u[1] = p[1];
  return r.v;
}

__device__ __forceinline__ v8f wmma_bf16(v16bf a, v16bf b, v8f c) {
  return __builtin_amdgcn_wmma_f32_16x16x32_bf16(false, a, false, b, (short)0, c, false, false);
}

// ---------- weight packing ----------
// B layout (wave32): lane holds column n = lane&15; element i -> K = kt*32 + 8*(lane>>4) + (i&7) + 16*(i>>3).
__global__ void pack_w_kernel(const float* __restrict__ W, __bf16* __restrict__ out) {
  int frag = blockIdx.x;  // (K/32)*4 fragments
  int kt = frag >> 2, nt = frag & 3;
  int lane = threadIdx.x & 31;
  int half = lane >> 4;
  int col = nt * 16 + (lane & 15);
#pragma unroll
  for (int i = 0; i < 16; ++i) {
    int k = kt * 32 + half * 8 + (i & 7) + 16 * (i >> 3);
    out[frag * 512 + lane * 16 + i] = f2bf(W[k * DD + col]);
  }
}

// ---------- H = relu(X @ W64 + b), full 16-row tiles only, branchless ----------
template <bool HOUT>
__global__ void encode64_kernel(const float* __restrict__ X, const __bf16* __restrict__ Wp,
                                const float* __restrict__ bias, __bf16* __restrict__ Hbf,
                                float* __restrict__ Hout, int fullTiles) {
  __shared__ __align__(16) __bf16 sW[8 * 512];
  stage_w(Wp, sW, 8);
  int wave = blockIdx.x * (blockDim.x >> 5) + (threadIdx.x >> 5);
  if (wave >= fullTiles) return;
  const int r0   = wave * 16;
  const int lane = threadIdx.x & 31;
  const int half = lane >> 4;
  const int nc   = lane & 15;
  const unsigned aoff = ((unsigned)(r0 + nc)) << 6;
  v16bf a0 = load_a_f32(X, aoff, 0, half);
  v16bf a1 = load_a_f32(X, aoff, 32, half);
#pragma unroll
  for (int nt = 0; nt < 4; ++nt) {
    float bv = bias[nt * 16 + nc];
    v8f acc;
#pragma unroll
    for (int g = 0; g < 8; ++g) acc[g] = bv;
    acc = wmma_bf16(a0, load_b_lds(sW, nt, lane), acc);
    acc = wmma_bf16(a1, load_b_lds(sW, 4 + nt, lane), acc);
#pragma unroll
    for (int g = 0; g < 8; ++g) {
      unsigned o = (((unsigned)(r0 + g + 8 * half)) << 6) + nt * 16 + nc;
      float v = relu(acc[g]);
      Hbf[o] = f2bf(v);
      if (HOUT) Hout[o] = v;
    }
  }
}

// encode tail (rows % 16), plain per-element dot; launched only when needed.
template <bool HOUT>
__global__ void encode_tail_kernel(const float* __restrict__ X, const float* __restrict__ W,
                                   const float* __restrict__ bias, __bf16* __restrict__ Hbf,
                                   float* __restrict__ Hout, int r0, int rem) {
  int t = threadIdx.x;
  if (t >= rem * DD) return;
  int m = r0 + (t >> 6), n = t & 63;
  float s = bias[n];
  for (int k = 0; k < DD; ++k) s += bf2f(f2bf(X[m * DD + k])) * bf2f(f2bf(W[k * DD + n]));
  float v = relu(s);
  unsigned o = (((unsigned)m) << 6) + n;
  Hbf[o] = f2bf(v);
  if (HOUT) Hout[o] = v;
}

// ---------- m = relu(Hbf[gidx] @ W64 + b); AGG[sidx] += m ----------
// Branchless scatter: out-of-range lanes scatter into a dump row at index nAgg.
__global__ void msg_scatter_kernel(const __bf16* __restrict__ H, const int* __restrict__ gidx,
                                   const int* __restrict__ sidx, const __bf16* __restrict__ Wp,
                                   const float* __restrict__ bias, float* __restrict__ AGG,
                                   int E, int nAgg) {
  __shared__ __align__(16) __bf16 sW[8 * 512];
  stage_w(Wp, sW, 8);
  const int lane = threadIdx.x & 31;
  const int half = lane >> 4;
  const int nc   = lane & 15;
  float bv[4];
#pragma unroll
  for (int nt = 0; nt < 4; ++nt) bv[nt] = bias[nt * 16 + nc];
  const unsigned dumpOff = (((unsigned)nAgg) << 6) + nc;

  const int ntiles = (E + 15) >> 4;
  const int nwaves = gridDim.x * (blockDim.x >> 5);
  for (int tile = blockIdx.x * (blockDim.x >> 5) + (threadIdx.x >> 5); tile < ntiles;
       tile += nwaves) {
    const int base = tile * 16;
    unsigned offs[8];
    if (base + 16 <= E) {
      const int4* sp = (const int4*)(sidx + base + 8 * half);
      int4 d0 = sp[0], d1 = sp[1];
      int ds[8] = {d0.x, d0.y, d0.z, d0.w, d1.x, d1.y, d1.z, d1.w};
#pragma unroll
      for (int g = 0; g < 8; ++g) offs[g] = (((unsigned)ds[g]) << 6) + nc;
    } else {
#pragma unroll
      for (int g = 0; g < 8; ++g) {
        int em = base + g + 8 * half;
        int emc = em < E ? em : E - 1;
        unsigned real = (((unsigned)sidx[emc]) << 6) + nc;
        offs[g] = (em < E) ? real : dumpOff;
      }
    }
    int e = base + nc;
    e = e < E ? e : E - 1;
    const unsigned aoff = ((unsigned)gidx[e]) << 6;
    v16bf a0 = load_a_bf(H, aoff, 0, half);
    v16bf a1 = load_a_bf(H, aoff, 32, half);
#pragma unroll
    for (int nt = 0; nt < 4; ++nt) {
      v8f c;
#pragma unroll
      for (int g = 0; g < 8; ++g) c[g] = bv[nt];
      c = wmma_bf16(a0, load_b_lds(sW, nt, lane), c);
      c = wmma_bf16(a1, load_b_lds(sW, 4 + nt, lane), c);
#pragma unroll
      for (int g = 0; g < 8; ++g)
        atomicAdd(AGG + offs[g] + nt * 16, relu(c[g]));
    }
  }
}

// ---------- H = relu(concat([H, G]) @ W128 + b), in place, full tiles, branchless ----------
template <bool HOUT>
__global__ void update128_kernel(__bf16* __restrict__ Hbf, const float* __restrict__ G,
                                 const __bf16* __restrict__ Wp, const float* __restrict__ bias,
                                 float* __restrict__ Hout, int fullTiles) {
  __shared__ __align__(16) __bf16 sW[16 * 512];
  stage_w(Wp, sW, 16);
  int wave = blockIdx.x * (blockDim.x >> 5) + (threadIdx.x >> 5);
  if (wave >= fullTiles) return;
  const int r0   = wave * 16;
  const int lane = threadIdx.x & 31;
  const int half = lane >> 4;
  const int nc   = lane & 15;
  const unsigned aoff = ((unsigned)(r0 + nc)) << 6;
  v16bf a0 = load_a_bf(Hbf, aoff, 0, half);
  v16bf a1 = load_a_bf(Hbf, aoff, 32, half);
  v16bf a2 = load_a_f32(G, aoff, 0, half);
  v16bf a3 = load_a_f32(G, aoff, 32, half);
#pragma unroll
  for (int nt = 0; nt < 4; ++nt) {
    float bv = bias[nt * 16 + nc];
    v8f acc;
#pragma unroll
    for (int g = 0; g < 8; ++g) acc[g] = bv;
    acc = wmma_bf16(a0, load_b_lds(sW, nt, lane), acc);
    acc = wmma_bf16(a1, load_b_lds(sW, 4 + nt, lane), acc);
    acc = wmma_bf16(a2, load_b_lds(sW, 8 + nt, lane), acc);
    acc = wmma_bf16(a3, load_b_lds(sW, 12 + nt, lane), acc);
#pragma unroll
    for (int g = 0; g < 8; ++g) {
      unsigned o = (((unsigned)(r0 + g + 8 * half)) << 6) + nt * 16 + nc;
      float v = relu(acc[g]);
      Hbf[o] = f2bf(v);
      if (HOUT) Hout[o] = v;
    }
  }
}

// update tail: race-free via LDS snapshot; launched only when rows % 16 != 0.
template <bool HOUT>
__global__ void update_tail_kernel(__bf16* __restrict__ Hbf, const float* __restrict__ G,
                                   const float* __restrict__ W, const float* __restrict__ bias,
                                   float* __restrict__ Hout, int r0, int rem) {
  __shared__ float sA[15 * 128];
  int t = threadIdx.x;
  int total = rem * 128;
  for (int i = t; i < total; i += blockDim.x) {
    int mm = i >> 7, k = i & 127;
    sA[i] = (k < DD) ? bf2f(Hbf[(r0 + mm) * DD + k])
                     : bf2f(f2bf(G[(r0 + mm) * DD + (k - DD)]));
  }
  __syncthreads();
  if (t < rem * DD) {
    int m = t >> 6, n = t & 63;
    float s = bias[n];
    for (int k = 0; k < 2 * DD; ++k) s += sA[m * 128 + k] * bf2f(f2bf(W[k * DD + n]));
    float v = relu(s);
    unsigned o = (((unsigned)(r0 + m)) << 6) + n;
    Hbf[o] = f2bf(v);
    if (HOUT) Hout[o] = v;
  }
}

// ---------- host orchestration ----------

extern "C" void kernel_launch(void* const* d_in, const int* in_sizes, int n_in,
                              void* d_out, int out_size, void* d_ws, size_t ws_size,
                              hipStream_t stream) {
  (void)n_in; (void)out_size; (void)ws_size;
  const float* x_src    = (const float*)d_in[0];
  const float* x_dst    = (const float*)d_in[1];
  const int*   src_idx  = (const int*)d_in[2];
  const int*   dst_idx  = (const int*)d_in[3];
  const float* W_in_src = (const float*)d_in[4];
  const float* b_in_src = (const float*)d_in[5];
  const float* W_in_dst = (const float*)d_in[6];
  const float* b_in_dst = (const float*)d_in[7];
  const float* W_msg_sd = (const float*)d_in[8];
  const float* b_msg_sd = (const float*)d_in[9];
  const float* W_upd_dst= (const float*)d_in[10];
  const float* b_upd_dst= (const float*)d_in[11];
  const float* W_msg_ds = (const float*)d_in[12];
  const float* b_msg_ds = (const float*)d_in[13];
  const float* W_upd_src= (const float*)d_in[14];
  const float* b_upd_src= (const float*)d_in[15];

  const int Nsrc = in_sizes[0] / DD;
  const int Ndst = in_sizes[1] / DD;
  const int E    = in_sizes[2];
  const int L    = in_sizes[8] / (DD * DD);

  char* ws = (char*)d_ws;
  float*  agg_s = (float*)ws;   ws += (size_t)(Nsrc + 1) * DD * sizeof(float);  // +1 dump row
  float*  agg_d = (float*)ws;   ws += (size_t)(Ndst + 1) * DD * sizeof(float);  // +1 dump row
  __bf16* h_s   = (__bf16*)ws;  ws += (size_t)Nsrc * DD * sizeof(__bf16);
  __bf16* h_d   = (__bf16*)ws;  ws += (size_t)Ndst * DD * sizeof(__bf16);
  const size_t F64  = 8  * 512;
  const size_t F128 = 16 * 512;
  __bf16* pWin_s   = (__bf16*)ws; ws += F64 * sizeof(__bf16);
  __bf16* pWin_d   = (__bf16*)ws; ws += F64 * sizeof(__bf16);
  __bf16* pWmsg_sd = (__bf16*)ws; ws += (size_t)L * F64  * sizeof(__bf16);
  __bf16* pWmsg_ds = (__bf16*)ws; ws += (size_t)L * F64  * sizeof(__bf16);
  __bf16* pWupd_d  = (__bf16*)ws; ws += (size_t)L * F128 * sizeof(__bf16);
  __bf16* pWupd_s  = (__bf16*)ws; ws += (size_t)L * F128 * sizeof(__bf16);

  float* out_f32 = (float*)d_out;

  // 1) pack weights
  pack_w_kernel<<<8, 32, 0, stream>>>(W_in_src, pWin_s);
  pack_w_kernel<<<8, 32, 0, stream>>>(W_in_dst, pWin_d);
  for (int l = 0; l < L; ++l) {
    pack_w_kernel<<<8, 32, 0, stream>>>(W_msg_sd + (size_t)l * DD * DD, pWmsg_sd + (size_t)l * F64);
    pack_w_kernel<<<8, 32, 0, stream>>>(W_msg_ds + (size_t)l * DD * DD, pWmsg_ds + (size_t)l * F64);
    pack_w_kernel<<<16, 32, 0, stream>>>(W_upd_dst + (size_t)l * 2 * DD * DD, pWupd_d + (size_t)l * F128);
    pack_w_kernel<<<16, 32, 0, stream>>>(W_upd_src + (size_t)l * 2 * DD * DD, pWupd_s + (size_t)l * F128);
  }

  const int WPB = 8;  // waves per 256-thread block
  auto blocks = [&](int t) { return (t + WPB - 1) / WPB; };
  const int ftS = Nsrc >> 4, remS = Nsrc & 15;
  const int ftD = Ndst >> 4, remD = Ndst & 15;

  // 2) encoders (full tiles via WMMA, remainder via tail kernel)
  if (ftS) encode64_kernel<false><<<blocks(ftS), 256, 0, stream>>>(x_src, pWin_s, b_in_src, h_s, nullptr, ftS);
  if (remS) encode_tail_kernel<false><<<1, 1024, 0, stream>>>(x_src, W_in_src, b_in_src, h_s, nullptr, ftS * 16, remS);
  if (ftD) encode64_kernel<true><<<blocks(ftD), 256, 0, stream>>>(x_dst, pWin_d, b_in_dst, h_d, out_f32, ftD);
  if (remD) encode_tail_kernel<true><<<1, 1024, 0, stream>>>(x_dst, W_in_dst, b_in_dst, h_d, out_f32, ftD * 16, remD);

  // persistent grid for the edge kernels
  int msg_blocks = 1024;
  int et = (E + 15) >> 4;
  if (blocks(et) < msg_blocks) msg_blocks = blocks(et) > 0 ? blocks(et) : 1;

  // 3) message-passing layers
  for (int l = 0; l < L; ++l) {
    hipMemsetAsync(agg_d, 0, (size_t)Ndst * DD * sizeof(float), stream);
    msg_scatter_kernel<<<msg_blocks, 256, 0, stream>>>(
        h_s, src_idx, dst_idx, pWmsg_sd + (size_t)l * F64, b_msg_sd + (size_t)l * DD, agg_d, E, Ndst);
    if (ftD) update128_kernel<true><<<blocks(ftD), 256, 0, stream>>>(
        h_d, agg_d, pWupd_d + (size_t)l * F128, b_upd_dst + (size_t)l * DD, out_f32, ftD);
    if (remD) update_tail_kernel<true><<<1, 1024, 0, stream>>>(
        h_d, agg_d, W_upd_dst + (size_t)l * 2 * DD * DD, b_upd_dst + (size_t)l * DD, out_f32, ftD * 16, remD);

    hipMemsetAsync(agg_s, 0, (size_t)Nsrc * DD * sizeof(float), stream);
    msg_scatter_kernel<<<msg_blocks, 256, 0, stream>>>(
        h_d, dst_idx, src_idx, pWmsg_ds + (size_t)l * F64, b_msg_ds + (size_t)l * DD, agg_s, E, Nsrc);
    if (ftS) update128_kernel<false><<<blocks(ftS), 256, 0, stream>>>(
        h_s, agg_s, pWupd_s + (size_t)l * F128, b_upd_src + (size_t)l * DD, nullptr, ftS);
    if (remS) update_tail_kernel<false><<<1, 1024, 0, stream>>>(
        h_s, agg_s, W_upd_src + (size_t)l * 2 * DD * DD, b_upd_src + (size_t)l * DD, nullptr, ftS * 16, remS);
  }
  // d_out holds the f32 h_d from the final dst update
}
